// QVI_inter_Ds_63763084476661
// MI455X (gfx1250) — compile-verified, matrix operand order checked by reading, not run on hardware
//
#include <hip/hip_runtime.h>
#include <hip/hip_bf16.h>
#include <math.h>

// ---------------------------------------------------------------------------
// QVI frame interpolation for MI455X (gfx1250, wave32).
// Convs: implicit-GEMM on v_wmma_f32_16x16x32_f16 (fp16 in, f32 acc),
//        LDS-staged B tiles pre-swizzled to the WMMA lane layout, shared by
//        up to 4 Cout-tile waves per block.  KSZ templated so tap index math
//        strength-reduces (no runtime integer division in the hot loop).
// Everything else: bandwidth-bound flat kernels writing straight into
// channel slices of the concat buffers (no concatenation copies).
// ---------------------------------------------------------------------------

typedef __attribute__((ext_vector_type(16))) _Float16 v16h;
typedef __attribute__((ext_vector_type(8)))  _Float16 v8h;
typedef __attribute__((ext_vector_type(8)))  float    v8f;

#define ACT_NONE 0
#define ACT_LRELU 1
#define ACT_SIG 2

// ============================ WMMA implicit-GEMM conv ======================
// Block = 4 waves.  MW waves share one 16-pixel tile (different Cout tiles);
// NW = 4/MW pixel-tile groups per block.  B tile (32K x 16N fp16) is staged
// to LDS once per group in WMMA register layout; consume = 32B LDS load.
template <int KSZ, int MW>
__global__ __launch_bounds__(128)
void k_conv_wmma(const float* __restrict__ in, int inCtot, int inC0, int Cin,
                 int Hin, int Win,
                 float* __restrict__ out, int outCtot, int outC0, int Cout,
                 int Hout, int Wout,
                 const _Float16* __restrict__ Wh, const float* __restrict__ bias,
                 int Kpad, int pad, int stride, int act, int Nimg)
{
  constexpr int KK = KSZ * KSZ;
  constexpr int NW = 4 / MW;                 // pixel-tile groups per block
  const int lane = threadIdx.x;              // 0..31
  const int w    = threadIdx.y;              // 0..3
  const int g    = w / MW;                   // pixel-tile group
  const int wm   = w % MW;                   // Cout tile within group

  const int HWo  = Hout * Wout;
  const int Npix = Nimg * HWo;
  const int Nbase = (blockIdx.x * NW + g) * 16;
  const int Mbase = (blockIdx.y * MW + wm) * 16;
  const int lo = lane & 15;
  const int hi = lane >> 4;                  // 0 or 1
  const int koff = hi * 8;

  __shared__ __align__(32) _Float16 sB[NW][2][512];

  // This lane's pixel (same for staging and consuming: column lo of the tile)
  int p = Nbase + lo; if (p >= Npix) p = Npix - 1;
  const int img = p / HWo;
  const int pr  = p - img * HWo;
  const int oy  = pr / Wout;
  const int ox  = pr - oy * Wout;
  const int iy0 = oy * stride - pad;
  const int ix0 = ox * stride - pad;
  const size_t plane = (size_t)Hin * (size_t)Win;
  const float* inImg = in + (size_t)(img * inCtot + inC0) * plane;

  // This lane's A row (weights): two contiguous 16B chunks per K-step
  const _Float16* Arow = Wh + (size_t)(Mbase + lo) * (size_t)Kpad;
  const int CK = Cin * KK;

  v8f acc = {};
  int buf = 0;
  for (int kk = 0; kk < Kpad; kk += 32, buf ^= 1) {
    if (wm == 0) {
      // stage 32(K) x 16(N) tile, pre-swizzled to the WMMA lane layout
      const int khalf = hi << 4;             // lanes 0-15: K 0..15, 16-31: 16..31
#pragma unroll
      for (int j = 0; j < 16; ++j) {
        int kl = khalf + j;
        int k  = kk + kl;
        float v = 0.f;
        if (k < CK) {
          int c  = k / KK;                   // constexpr KK -> mul/shift
          int r  = k - c * KK;
          int ky = r / KSZ;
          int kx = r - ky * KSZ;
          int iy = iy0 + ky, ix = ix0 + kx;
          if (iy >= 0 && iy < Hin && ix >= 0 && ix < Win)
            v = inImg[(size_t)c * plane + (size_t)iy * Win + ix];
        }
        int dl = lo + (((kl >> 3) & 1) << 4);
        int di = (kl & 7) + ((kl >> 4) << 3);
        sB[g][buf][dl * 16 + di] = (_Float16)v;
      }
    }
    __builtin_prefetch(Arow + kk + 32, 0, 1);          // global_prefetch
    v8h a0 = *(const v8h*)(Arow + kk + koff);
    v8h a1 = *(const v8h*)(Arow + kk + koff + 16);
    v16h A;
#pragma unroll
    for (int i = 0; i < 8; ++i) { A[i] = a0[i]; A[i + 8] = a1[i]; }
    __syncthreads();                                   // stage(i) -> read(i)
    v16h B = *(const v16h*)&sB[g][buf][lane * 16];     // ds_load_b128 x2
    acc = __builtin_amdgcn_wmma_f32_16x16x32_f16(false, A, false, B,
                                                 (short)0, acc, false, false);
  }

  const int pst = Nbase + lo;
  if (pst < Npix) {
    const int img2 = pst / HWo;
    const int pr2  = pst - img2 * HWo;
#pragma unroll
    for (int r = 0; r < 8; ++r) {
      int cg = Mbase + r + hi * 8;                     // lanes>=16 -> M+8
      if (cg < Cout) {
        float v = acc[r] + bias[cg];
        if (act == ACT_LRELU)     v = v >= 0.f ? v : 0.1f * v;
        else if (act == ACT_SIG)  v = 1.f / (1.f + expf(-v));
        out[(size_t)(img2 * outCtot + outC0 + cg) * HWo + pr2] = v;
      }
    }
  }
}

// ============================ weight packing ===============================
__global__ void k_pack_w(const float* __restrict__ src, _Float16* __restrict__ dst,
                         int O, int CK, int Opad, int Kpad)
{
  int idx = blockIdx.x * blockDim.x + threadIdx.x;
  if (idx >= Opad * Kpad) return;
  int o = idx / Kpad, k = idx - o * Kpad;
  float v = (o < O && k < CK) ? src[o * CK + k] : 0.f;
  dst[idx] = (_Float16)v;
}

// ============================ elementwise kernels ==========================
__global__ void k_zero(float* p, int n)
{
  int i = blockIdx.x * blockDim.x + threadIdx.x;
  if (i < n) p[i] = 0.f;
}

// quadratic flow fusion + nearest /3 downsample fused
__global__ void k_quad(const float* __restrict__ Fa, const float* __restrict__ Fb,
                       const float* __restrict__ tp, int tsel,
                       float* __restrict__ out, int N, int H2, int W2, int Hd, int Wd)
{
  int HW2 = H2 * W2; int tot = N * 2 * HW2;
  int i = blockIdx.x * blockDim.x + threadIdx.x; if (i >= tot) return;
  int n = i / (2 * HW2); int r = i - n * 2 * HW2; int c = r / HW2; int pix = r - c * HW2;
  int y = pix / W2, x = pix - y * W2;
  int ys = 3 * y + 1, xs = 3 * x + 1;
  size_t si = (size_t)(n * 2 + c) * ((size_t)Hd * Wd) + (size_t)ys * Wd + xs;
  float t = tp[0]; if (tsel) t = 1.f - t;
  float F = 0.5f * ((t + t * t) * Fa[si] - (t - t * t) * Fb[si]);
  out[i] = F * (1.f / 3.f);
}

// forward bilinear splat (FlowReversal); img == flow here
__global__ void k_fwarp(const float* __restrict__ f2, float* buf, float* nrm,
                        int N, int H, int W)
{
  int HW = H * W;
  int i = blockIdx.x * blockDim.x + threadIdx.x; if (i >= N * HW) return;
  int n = i / HW, pix = i - n * HW, y = pix / W, x = pix - y * W;
  float u = f2[(n * 2 + 0) * HW + pix], v = f2[(n * 2 + 1) * HW + pix];
  float xx = x + u, yy = y + v;
  int x0 = (int)floorf(xx), y0 = (int)floorf(yy);
  for (int dy = 0; dy < 2; ++dy)
    for (int dx = 0; dx < 2; ++dx) {
      int xi = x0 + dx, yi = y0 + dy;
      if (xi < 0 || xi >= W || yi < 0 || yi >= H) continue;
      float w = (1.f - fabsf(xx - xi)) * (1.f - fabsf(yy - yi));
      int d = yi * W + xi;
      atomicAdd(&buf[(n * 2 + 0) * HW + d], w * u);
      atomicAdd(&buf[(n * 2 + 1) * HW + d], w * v);
      atomicAdd(&nrm[n * HW + d], w);
    }
}

__global__ void k_normfill(const float* buf, const float* nrm, const float* fillsrc,
                           float* ftA, float* hole, int N, int HW)
{
  int i = blockIdx.x * blockDim.x + threadIdx.x; if (i >= N * HW) return;
  int n = i / HW, pix = i - n * HW;
  float nv = nrm[i];
  bool h = (nv == 0.f);
  hole[i] = h ? 1.f : 0.f;
  for (int c = 0; c < 2; ++c) {
    float b  = buf[(n * 2 + c) * HW + pix];
    float ft = nv > 0.f ? -b / nv : -b;
    float fl = -fillsrc[(n * 2 + c) * HW + pix];
    ftA[(n * 2 + c) * HW + pix] = h ? fl : ft;
  }
}

__global__ void k_box7(const float* __restrict__ src, float* dst, int NC, int H, int W)
{
  int HW = H * W;
  int i = blockIdx.x * blockDim.x + threadIdx.x; if (i >= NC * HW) return;
  int nc = i / HW, pix = i - nc * HW, y = pix / W, x = pix - y * W;
  const float* S = src + (size_t)nc * HW;
  float s = 0.f;
  for (int dy = -3; dy <= 3; ++dy) {
    int yy = y + dy; if (yy < 0 || yy >= H) continue;
    for (int dx = -3; dx <= 3; ++dx) {
      int xx = x + dx; if (xx < 0 || xx >= W) continue;
      s += S[yy * W + xx];
    }
  }
  dst[i] = s * (1.f / 49.f);
}

// generic bilinear resize on channel slices, with value scale
__global__ void k_resize(const float* __restrict__ src, int sCt, int sC0, int Hi, int Wi,
                         float* __restrict__ dst, int dCt, int dC0, int Ho, int Wo,
                         int C, int N, float vscale)
{
  long tot = (long)N * C * Ho * Wo;
  long i = (long)blockIdx.x * blockDim.x + threadIdx.x; if (i >= tot) return;
  int HWo = Ho * Wo;
  int n = (int)(i / ((long)C * HWo));
  int r = (int)(i - (long)n * C * HWo);
  int c = r / HWo; int pix = r - c * HWo;
  int y = pix / Wo, x = pix - y * Wo;
  float sy = (y + 0.5f) * (float)Hi / (float)Ho - 0.5f;
  float sx = (x + 0.5f) * (float)Wi / (float)Wo - 0.5f;
  sy = fminf(fmaxf(sy, 0.f), (float)(Hi - 1));
  sx = fminf(fmaxf(sx, 0.f), (float)(Wi - 1));
  int y0 = (int)sy, x0 = (int)sx;
  int y1 = min(y0 + 1, Hi - 1), x1 = min(x0 + 1, Wi - 1);
  float wy = sy - y0, wx = sx - x0;
  const float* S = src + (size_t)(n * sCt + sC0 + c) * ((size_t)Hi * Wi);
  float v = (1 - wy) * ((1 - wx) * S[y0 * Wi + x0] + wx * S[y0 * Wi + x1]) +
            wy       * ((1 - wx) * S[y1 * Wi + x0] + wx * S[y1 * Wi + x1]);
  dst[(size_t)(n * dCt + dC0 + c) * HWo + pix] = v * vscale;
}

__global__ void k_select(const float* ftA, const float* up, const float* hole,
                         float* dst, int N, int HW)
{
  int tot = N * 2 * HW;
  int i = blockIdx.x * blockDim.x + threadIdx.x; if (i >= tot) return;
  int n = i / (2 * HW); int r = i - n * 2 * HW; int pix = r % HW;
  dst[i] = hole[n * HW + pix] > 0.5f ? up[i] : ftA[i];
}

// nearest x3 upsample with *3 flow rescale, into a channel slice
__global__ void k_up3(const float* __restrict__ src, float* __restrict__ dst,
                      int dCt, int dC0, int N, int H2, int W2, int Hd, int Wd)
{
  int HWd = Hd * Wd; long tot = (long)N * 2 * HWd;
  long i = (long)blockIdx.x * blockDim.x + threadIdx.x; if (i >= tot) return;
  int n = (int)(i / (2 * (long)HWd)); int r = (int)(i - (long)n * 2 * HWd);
  int c = r / HWd; int pix = r - c * HWd;
  int y = pix / Wd, x = pix - y * Wd;
  float v = src[(size_t)(n * 2 + c) * (H2 * W2) + (y / 3) * W2 + (x / 3)];
  dst[(size_t)(n * dCt + dC0 + c) * HWd + pix] = v * 3.f;
}

__global__ void k_copy_slice(const float* __restrict__ src, int sCt, int sC0,
                             float* __restrict__ dst, int dCt, int dC0,
                             int C, int N, int HW)
{
  long tot = (long)N * C * HW;
  long i = (long)blockIdx.x * blockDim.x + threadIdx.x; if (i >= tot) return;
  int n = (int)(i / ((long)C * HW)); int r = (int)(i - (long)n * C * HW);
  int c = r / HW; int pix = r - c * HW;
  dst[(size_t)(n * dCt + dC0 + c) * HW + pix] =
      src[(size_t)(n * sCt + sC0 + c) * HW + pix];
}

// border-clamped bilinear backward warp on channel slices
__global__ void k_backwarp(const float* __restrict__ img, int iCt, int iC0, int C,
                           const float* __restrict__ fl, int fCt, int fC0,
                           float* __restrict__ out, int oCt, int oC0,
                           int N, int H, int W)
{
  int HW = H * W; long tot = (long)N * C * HW;
  long i = (long)blockIdx.x * blockDim.x + threadIdx.x; if (i >= tot) return;
  int n = (int)(i / ((long)C * HW)); int r = (int)(i - (long)n * C * HW);
  int c = r / HW; int pix = r - c * HW;
  int y = pix / W, x = pix - y * W;
  float fx = fl[(size_t)(n * fCt + fC0 + 0) * HW + pix];
  float fy = fl[(size_t)(n * fCt + fC0 + 1) * HW + pix];
  float xc = fminf(fmaxf(x + fx, 0.f), (float)(W - 1));
  float yc = fminf(fmaxf(y + fy, 0.f), (float)(H - 1));
  int x0 = (int)floorf(xc), y0 = (int)floorf(yc);
  int x1 = min(x0 + 1, W - 1), y1 = min(y0 + 1, H - 1);
  float wx = xc - x0, wy = yc - y0;
  const float* S = img + (size_t)(n * iCt + iC0 + c) * HW;
  float v = (1 - wy) * ((1 - wx) * S[y0 * W + x0] + wx * S[y0 * W + x1]) +
            wy       * ((1 - wx) * S[y1 * W + x0] + wx * S[y1 * W + x1]);
  out[(size_t)(n * oCt + oC0 + c) * HW + pix] = v;
}

// Ftr = backwarp(Ft, 10*tanh(out[cFlow:cFlow+2])) + out[cAdd:cAdd+2]
__global__ void k_refine(const float* __restrict__ ft, int ftCt, int ftC0,
                         const float* __restrict__ o8, int cFlow, int cAdd,
                         float* __restrict__ dst, int N, int H, int W)
{
  int HW = H * W; long tot = (long)N * 2 * HW;
  long i = (long)blockIdx.x * blockDim.x + threadIdx.x; if (i >= tot) return;
  int n = (int)(i / (2 * (long)HW)); int r = (int)(i - (long)n * 2 * HW);
  int c = r / HW; int pix = r - c * HW;
  int y = pix / W, x = pix - y * W;
  float dx = 10.f * tanhf(o8[(size_t)(n * 8 + cFlow + 0) * HW + pix]);
  float dy = 10.f * tanhf(o8[(size_t)(n * 8 + cFlow + 1) * HW + pix]);
  float xc = fminf(fmaxf(x + dx, 0.f), (float)(W - 1));
  float yc = fminf(fmaxf(y + dy, 0.f), (float)(H - 1));
  int x0 = (int)floorf(xc), y0 = (int)floorf(yc);
  int x1 = min(x0 + 1, W - 1), y1 = min(y0 + 1, H - 1);
  float wx = xc - x0, wy = yc - y0;
  const float* S = ft + (size_t)(n * ftCt + ftC0 + c) * HW;
  float v = (1 - wy) * ((1 - wx) * S[y0 * W + x0] + wx * S[y0 * W + x1]) +
            wy       * ((1 - wx) * S[y1 * W + x0] + wx * S[y1 * W + x1]);
  dst[(size_t)(n * 2 + c) * HW + pix] = v + o8[(size_t)(n * 8 + cAdd + c) * HW + pix];
}

// central-difference gradient magnitude (zero pad)
__global__ void k_grad(const float* __restrict__ src, int sCt, int sC0,
                       float* __restrict__ dst, int dCt, int dC0,
                       int C, int N, int H, int W)
{
  int HW = H * W; long tot = (long)N * C * HW;
  long i = (long)blockIdx.x * blockDim.x + threadIdx.x; if (i >= tot) return;
  int n = (int)(i / ((long)C * HW)); int r = (int)(i - (long)n * C * HW);
  int c = r / HW; int pix = r - c * HW;
  int y = pix / W, x = pix - y * W;
  const float* S = src + (size_t)(n * sCt + sC0 + c) * HW;
  float up = y > 0     ? S[(y - 1) * W + x] : 0.f;
  float dn = y < H - 1 ? S[(y + 1) * W + x] : 0.f;
  float lf = x > 0     ? S[y * W + x - 1]   : 0.f;
  float rt = x < W - 1 ? S[y * W + x + 1]   : 0.f;
  float gv = dn - up, gh = rt - lf;
  dst[(size_t)(n * dCt + dC0 + c) * HW + pix] = sqrtf(gv * gv + gh * gh + 1e-6f);
}

// 5x5 zero-pad max ("erode" in the reference)
__global__ void k_erode(const float* __restrict__ src, float* dst, int N, int H, int W)
{
  int HW = H * W;
  int i = blockIdx.x * blockDim.x + threadIdx.x; if (i >= N * HW) return;
  int n = i / HW, pix = i - n * HW, y = pix / W, x = pix - y * W;
  float m = -INFINITY;
  for (int dy = -2; dy <= 2; ++dy)
    for (int dx = -2; dx <= 2; ++dx) {
      int yy = y + dy, xx = x + dx;
      float v = (yy >= 0 && yy < H && xx >= 0 && xx < W) ? src[n * HW + yy * W + xx] : 0.f;
      m = fmaxf(m, v);
    }
  dst[i] = m;
}

__global__ void k_final(const float* __restrict__ I1, const float* __restrict__ I2,
                        const float* __restrict__ I1w, const float* __restrict__ I2w,
                        const float* __restrict__ Mf, const float* __restrict__ Msf,
                        const float* __restrict__ tp,
                        float* __restrict__ out, int N, int HW)
{
  long tot = (long)N * 3 * HW;
  long i = (long)blockIdx.x * blockDim.x + threadIdx.x; if (i >= tot) return;
  int n = (int)(i / (3 * (long)HW)); int r = (int)(i - (long)n * 3 * HW);
  int pix = r % HW;
  float t = tp[0];
  float m  = Mf[(size_t)n * HW + pix];
  float ms = Msf[(size_t)n * HW + pix];
  float a = (1.f - t) * m, b = t * (1.f - m);
  float itw = (a * I1w[i] + b * I2w[i]) / (a + b);
  float its = (1.f - t) * I1[i] + t * I2[i];
  out[i] = (1.f - ms) * itw + ms * its;
}

// ================================ host side ================================
extern "C" void kernel_launch(void* const* d_in, const int* in_sizes, int n_in,
                              void* d_out, int out_size, void* d_ws, size_t ws_size,
                              hipStream_t stream)
{
  (void)in_sizes; (void)n_in; (void)out_size; (void)ws_size;

  const int NIMG = 2;
  const int Hd = 384, Wd = 672;  const int HWd = Hd * Wd;
  const int Hh = 192, Wh2 = 336; const int HWh = Hh * Wh2;
  const int H2 = 128, W2 = 224;  const int HW2 = H2 * W2;
  const int H4 = 32,  W4 = 56;   const int HW4 = H4 * W4;
  const int Hf = 768, Wf = 1344; const int HWf = Hf * Wf;

  const float* F10 = (const float*)d_in[0];
  const float* F12 = (const float*)d_in[1];
  const float* F21 = (const float*)d_in[2];
  const float* F23 = (const float*)d_in[3];
  const float* I1D = (const float*)d_in[4];
  const float* I2D = (const float*)d_in[5];
  const float* I1F = (const float*)d_in[6];
  const float* I2F = (const float*)d_in[7];
  const float* tp  = (const float*)d_in[8];

  struct LayerCfg { int wIdx, bIdx, O, I, k; };
  const LayerCfg L[13] = {
    { 9, 10, 32, 20, 3}, {11, 12, 64, 32, 3}, {13, 14, 64, 64, 3},
    {15, 16, 32, 96, 3}, {17, 18,  8, 32, 3},
    {19, 20, 32, 38, 5}, {21, 22, 16, 32, 3}, {23, 24,  1, 16, 3},
    {25, 26, 32, 56, 3}, {27, 28, 32, 32, 3}, {29, 30, 16, 32, 3},
    {31, 32, 16, 16, 3}, {33, 34,  1, 16, 3}};

  // ---- bump allocator on workspace ----
  char* base = (char*)d_ws; size_t off = 0;
  auto allocF = [&](size_t n) -> float* {
    float* p = (float*)(base + off);
    off = (off + n * sizeof(float) + 255) & ~(size_t)255; return p; };
  auto allocH = [&](size_t n) -> _Float16* {
    _Float16* p = (_Float16*)(base + off);
    off = (off + n * sizeof(_Float16) + 255) & ~(size_t)255; return p; };

  dim3 TB(256);
  auto NB = [&](long n) { return dim3((unsigned)((n + 255) / 256)); };

  // ---- pack conv weights to fp16 [Opad][Kpad] ----
  _Float16* wh[13]; int Opad[13], Kpad[13];
  for (int l = 0; l < 13; ++l) {
    int CK = L[l].I * L[l].k * L[l].k;
    Opad[l] = (L[l].O + 15) & ~15;
    Kpad[l] = (CK + 31) & ~31;
    wh[l] = allocH((size_t)Opad[l] * Kpad[l]);
    k_pack_w<<<NB((long)Opad[l] * Kpad[l]), TB, 0, stream>>>(
        (const float*)d_in[L[l].wIdx], wh[l], L[l].O, CK, Opad[l], Kpad[l]);
  }

  auto conv = [&](int l, const float* in, int inCt, int inC0, int Cin, int Hin, int Win,
                  float* out, int outCt, int outC0, int Cout, int Hout, int Wout,
                  int pad, int stride, int act) {
    int Npix = NIMG * Hout * Wout;
    int MW = Opad[l] >= 64 ? 4 : (Opad[l] >= 32 ? 2 : 1);
    int NWg = 4 / MW;
    dim3 grid((unsigned)((Npix + 16 * NWg - 1) / (16 * NWg)),
              (unsigned)(Opad[l] / (16 * MW)));
    dim3 block(32, 4);
    const float* bi = (const float*)d_in[L[l].bIdx];
    if (L[l].k == 3) {
      if (MW == 4)
        k_conv_wmma<3, 4><<<grid, block, 0, stream>>>(in, inCt, inC0, Cin, Hin, Win,
            out, outCt, outC0, Cout, Hout, Wout, wh[l], bi, Kpad[l], pad, stride, act, NIMG);
      else if (MW == 2)
        k_conv_wmma<3, 2><<<grid, block, 0, stream>>>(in, inCt, inC0, Cin, Hin, Win,
            out, outCt, outC0, Cout, Hout, Wout, wh[l], bi, Kpad[l], pad, stride, act, NIMG);
      else
        k_conv_wmma<3, 1><<<grid, block, 0, stream>>>(in, inCt, inC0, Cin, Hin, Win,
            out, outCt, outC0, Cout, Hout, Wout, wh[l], bi, Kpad[l], pad, stride, act, NIMG);
    } else {
      if (MW == 2)
        k_conv_wmma<5, 2><<<grid, block, 0, stream>>>(in, inCt, inC0, Cin, Hin, Win,
            out, outCt, outC0, Cout, Hout, Wout, wh[l], bi, Kpad[l], pad, stride, act, NIMG);
      else
        k_conv_wmma<5, 1><<<grid, block, 0, stream>>>(in, inCt, inC0, Cin, Hin, Win,
            out, outCt, outC0, Cout, Hout, Wout, wh[l], bi, Kpad[l], pad, stride, act, NIMG);
    }
  };

  // ---- intermediate buffers ----
  const size_t NF2 = (size_t)NIMG * 2 * HW2, N1_2 = (size_t)NIMG * HW2;
  float *F1t2 = allocF(NF2), *F2t2 = allocF(NF2);
  float *b1 = allocF(NF2), *b2 = allocF(NF2);
  float *n1 = allocF(N1_2), *n2 = allocF(N1_2);
  float *ftA1 = allocF(NF2), *ftA2 = allocF(NF2);
  float *hole1 = allocF(N1_2), *hole2 = allocF(N1_2);
  float *box1 = allocF(NF2), *box2 = allocF(NF2);
  float *ds1 = allocF((size_t)NIMG * 2 * HW4), *ds2 = allocF((size_t)NIMG * 2 * HW4);
  float *up1 = allocF(NF2), *up2 = allocF(NF2);
  float *Ft1s = allocF(NF2), *Ft2s = allocF(NF2);

  float *U_in = allocF((size_t)NIMG * 20 * HWd);   // unet input (20ch)
  float *D_in = allocF((size_t)NIMG * 96 * HWd);   // [up(64), c1(32)]
  float *c2   = allocF((size_t)NIMG * 64 * HWh);
  float *c3   = allocF((size_t)NIMG * 64 * HWh);
  float *M_in = allocF((size_t)NIMG * 38 * HWd);   // [I1tf, I2tf, feat]
  float *S_in = allocF((size_t)NIMG * 56 * HWd);   // staticnet input
  float *out8 = allocF((size_t)NIMG * 8 * HWd);
  float *Ft1r = allocF((size_t)NIMG * 2 * HWd), *Ft2r = allocF((size_t)NIMG * 2 * HWd);
  float *mh1 = allocF((size_t)NIMG * 32 * HWd), *mh2 = allocF((size_t)NIMG * 16 * HWd);
  float *MDs = allocF((size_t)NIMG * HWd);
  float *sh1 = allocF((size_t)NIMG * 32 * HWd), *sh2 = allocF((size_t)NIMG * 32 * HWd);
  float *sh3 = allocF((size_t)NIMG * 16 * HWd), *sh4 = allocF((size_t)NIMG * 16 * HWd);
  float *s5  = allocF((size_t)NIMG * HWd);
  float *er1 = allocF((size_t)NIMG * HWd), *er2 = allocF((size_t)NIMG * HWd);
  float *Ftf1 = allocF((size_t)NIMG * 2 * HWf), *Ftf2 = allocF((size_t)NIMG * 2 * HWf);
  float *I1w = allocF((size_t)NIMG * 3 * HWf), *I2w = allocF((size_t)NIMG * 3 * HWf);
  float *Mf = allocF((size_t)NIMG * HWf), *Msf = allocF((size_t)NIMG * HWf);

  // ---- phase 1: quadratic flows, forward splat, hole fill ----
  k_quad<<<NB(NF2), TB, 0, stream>>>(F12, F10, tp, 0, F1t2, NIMG, H2, W2, Hd, Wd);
  k_quad<<<NB(NF2), TB, 0, stream>>>(F21, F23, tp, 1, F2t2, NIMG, H2, W2, Hd, Wd);
  k_zero<<<NB(NF2), TB, 0, stream>>>(b1, (int)NF2);
  k_zero<<<NB(NF2), TB, 0, stream>>>(b2, (int)NF2);
  k_zero<<<NB(N1_2), TB, 0, stream>>>(n1, (int)N1_2);
  k_zero<<<NB(N1_2), TB, 0, stream>>>(n2, (int)N1_2);
  k_fwarp<<<NB(N1_2), TB, 0, stream>>>(F1t2, b1, n1, NIMG, H2, W2);
  k_fwarp<<<NB(N1_2), TB, 0, stream>>>(F2t2, b2, n2, NIMG, H2, W2);
  k_normfill<<<NB(N1_2), TB, 0, stream>>>(b1, n1, F1t2, ftA1, hole1, NIMG, HW2);
  k_normfill<<<NB(N1_2), TB, 0, stream>>>(b2, n2, F2t2, ftA2, hole2, NIMG, HW2);
  k_box7<<<NB(NF2), TB, 0, stream>>>(ftA1, box1, NIMG * 2, H2, W2);
  k_box7<<<NB(NF2), TB, 0, stream>>>(ftA2, box2, NIMG * 2, H2, W2);
  k_resize<<<NB((long)NIMG * 2 * HW4), TB, 0, stream>>>(box1, 2, 0, H2, W2, ds1, 2, 0, H4, W4, 2, NIMG, 1.f);
  k_resize<<<NB((long)NIMG * 2 * HW4), TB, 0, stream>>>(box2, 2, 0, H2, W2, ds2, 2, 0, H4, W4, 2, NIMG, 1.f);
  k_resize<<<NB(NF2), TB, 0, stream>>>(ds1, 2, 0, H4, W4, up1, 2, 0, H2, W2, 2, NIMG, 1.f);
  k_resize<<<NB(NF2), TB, 0, stream>>>(ds2, 2, 0, H4, W4, up2, 2, 0, H2, W2, 2, NIMG, 1.f);
  k_select<<<NB(NF2), TB, 0, stream>>>(ftA1, up1, hole1, Ft1s, NIMG, HW2);
  k_select<<<NB(NF2), TB, 0, stream>>>(ftA2, up2, hole2, Ft2s, NIMG, HW2);
  k_up3<<<NB((long)NIMG * 2 * HWd), TB, 0, stream>>>(Ft1s, U_in, 20, 16, NIMG, H2, W2, Hd, Wd);
  k_up3<<<NB((long)NIMG * 2 * HWd), TB, 0, stream>>>(Ft2s, U_in, 20, 18, NIMG, H2, W2, Hd, Wd);

  // ---- phase 2: assemble unet input ----
  k_copy_slice<<<NB((long)NIMG * 3 * HWd), TB, 0, stream>>>(I1D, 3, 0, U_in, 20, 0, 3, NIMG, HWd);
  k_copy_slice<<<NB((long)NIMG * 3 * HWd), TB, 0, stream>>>(I2D, 3, 0, U_in, 20, 3, 3, NIMG, HWd);
  k_copy_slice<<<NB((long)NIMG * 2 * HWd), TB, 0, stream>>>(F12, 2, 0, U_in, 20, 12, 2, NIMG, HWd);
  k_copy_slice<<<NB((long)NIMG * 2 * HWd), TB, 0, stream>>>(F21, 2, 0, U_in, 20, 14, 2, NIMG, HWd);
  k_backwarp<<<NB((long)NIMG * 3 * HWd), TB, 0, stream>>>(I1D, 3, 0, 3, U_in, 20, 16, U_in, 20, 6, NIMG, Hd, Wd);
  k_backwarp<<<NB((long)NIMG * 3 * HWd), TB, 0, stream>>>(I2D, 3, 0, 3, U_in, 20, 18, U_in, 20, 9, NIMG, Hd, Wd);

  // ---- phase 3: unet (WMMA convs) ----
  conv(0, U_in, 20, 0, 20, Hd, Wd,  D_in, 96, 64, 32, Hd, Wd,  1, 1, ACT_LRELU);  // c1
  conv(1, D_in, 96, 64, 32, Hd, Wd, c2, 64, 0, 64, Hh, Wh2,    1, 2, ACT_LRELU);  // c2
  conv(2, c2, 64, 0, 64, Hh, Wh2,   c3, 64, 0, 64, Hh, Wh2,    1, 1, ACT_LRELU);  // c3
  k_resize<<<NB((long)NIMG * 64 * HWd), TB, 0, stream>>>(c3, 64, 0, Hh, Wh2, D_in, 96, 0, Hd, Wd, 64, NIMG, 1.f);
  conv(3, D_in, 96, 0, 96, Hd, Wd,  M_in, 38, 6, 32, Hd, Wd,   1, 1, ACT_LRELU);  // feat
  k_copy_slice<<<NB((long)NIMG * 32 * HWd), TB, 0, stream>>>(M_in, 38, 6, S_in, 56, 24, 32, NIMG, HWd);
  conv(4, M_in, 38, 6, 32, Hd, Wd,  out8, 8, 0, 8, Hd, Wd,     1, 1, ACT_NONE);   // out

  // ---- phase 4: flow refinement ----
  k_refine<<<NB((long)NIMG * 2 * HWd), TB, 0, stream>>>(U_in, 20, 16, out8, 4, 0, Ft1r, NIMG, Hd, Wd);
  k_refine<<<NB((long)NIMG * 2 * HWd), TB, 0, stream>>>(U_in, 20, 18, out8, 6, 2, Ft2r, NIMG, Hd, Wd);

  // ---- phase 5: warped images/gradients, staticnet input ----
  k_backwarp<<<NB((long)NIMG * 3 * HWd), TB, 0, stream>>>(I1D, 3, 0, 3, Ft1r, 2, 0, M_in, 38, 0, NIMG, Hd, Wd);
  k_backwarp<<<NB((long)NIMG * 3 * HWd), TB, 0, stream>>>(I2D, 3, 0, 3, Ft2r, 2, 0, M_in, 38, 3, NIMG, Hd, Wd);
  k_copy_slice<<<NB((long)NIMG * 6 * HWd), TB, 0, stream>>>(M_in, 38, 0, S_in, 56, 0, 6, NIMG, HWd);
  k_grad<<<NB((long)NIMG * 3 * HWd), TB, 0, stream>>>(I1D, 3, 0, S_in, 56, 18, 3, NIMG, Hd, Wd);
  k_grad<<<NB((long)NIMG * 3 * HWd), TB, 0, stream>>>(I2D, 3, 0, S_in, 56, 21, 3, NIMG, Hd, Wd);
  k_backwarp<<<NB((long)NIMG * 3 * HWd), TB, 0, stream>>>(S_in, 56, 18, 3, Ft1r, 2, 0, S_in, 56, 6, NIMG, Hd, Wd);
  k_backwarp<<<NB((long)NIMG * 3 * HWd), TB, 0, stream>>>(S_in, 56, 21, 3, Ft2r, 2, 0, S_in, 56, 9, NIMG, Hd, Wd);
  k_copy_slice<<<NB((long)NIMG * 3 * HWd), TB, 0, stream>>>(I1D, 3, 0, S_in, 56, 12, 3, NIMG, HWd);
  k_copy_slice<<<NB((long)NIMG * 3 * HWd), TB, 0, stream>>>(I2D, 3, 0, S_in, 56, 15, 3, NIMG, HWd);

  // ---- phase 6: masknet ----
  conv(5, M_in, 38, 0, 38, Hd, Wd, mh1, 32, 0, 32, Hd, Wd, 2, 1, ACT_LRELU);
  conv(6, mh1, 32, 0, 32, Hd, Wd,  mh2, 16, 0, 16, Hd, Wd, 1, 1, ACT_LRELU);
  conv(7, mh2, 16, 0, 16, Hd, Wd,  MDs, 1, 0, 1, Hd, Wd,   1, 1, ACT_SIG);

  // ---- phase 7: staticnet + erode ----
  conv(8,  S_in, 56, 0, 56, Hd, Wd, sh1, 32, 0, 32, Hd, Wd, 1, 1, ACT_LRELU);
  conv(9,  sh1, 32, 0, 32, Hd, Wd,  sh2, 32, 0, 32, Hd, Wd, 1, 1, ACT_LRELU);
  conv(10, sh2, 32, 0, 32, Hd, Wd,  sh3, 16, 0, 16, Hd, Wd, 1, 1, ACT_LRELU);
  conv(11, sh3, 16, 0, 16, Hd, Wd,  sh4, 16, 0, 16, Hd, Wd, 1, 1, ACT_LRELU);
  conv(12, sh4, 16, 0, 16, Hd, Wd,  s5, 1, 0, 1, Hd, Wd,    1, 1, ACT_SIG);
  k_erode<<<NB((long)NIMG * HWd), TB, 0, stream>>>(s5, er1, NIMG, Hd, Wd);
  k_erode<<<NB((long)NIMG * HWd), TB, 0, stream>>>(er1, er2, NIMG, Hd, Wd);

  // ---- phase 8: full-res composition ----
  k_resize<<<NB((long)NIMG * 2 * HWf), TB, 0, stream>>>(Ft1r, 2, 0, Hd, Wd, Ftf1, 2, 0, Hf, Wf, 2, NIMG, 2.f);
  k_resize<<<NB((long)NIMG * 2 * HWf), TB, 0, stream>>>(Ft2r, 2, 0, Hd, Wd, Ftf2, 2, 0, Hf, Wf, 2, NIMG, 2.f);
  k_backwarp<<<NB((long)NIMG * 3 * HWf), TB, 0, stream>>>(I1F, 3, 0, 3, Ftf1, 2, 0, I1w, 3, 0, NIMG, Hf, Wf);
  k_backwarp<<<NB((long)NIMG * 3 * HWf), TB, 0, stream>>>(I2F, 3, 0, 3, Ftf2, 2, 0, I2w, 3, 0, NIMG, Hf, Wf);
  k_resize<<<NB((long)NIMG * HWf), TB, 0, stream>>>(MDs, 1, 0, Hd, Wd, Mf, 1, 0, Hf, Wf, 1, NIMG, 1.f);
  k_resize<<<NB((long)NIMG * HWf), TB, 0, stream>>>(er2, 1, 0, Hd, Wd, Msf, 1, 0, Hf, Wf, 1, NIMG, 1.f);
  k_final<<<NB((long)NIMG * 3 * HWf), TB, 0, stream>>>(I1F, I2F, I1w, I2w, Mf, Msf, tp,
                                                       (float*)d_out, NIMG, HWf);
}
